// DoubleStreamBlock_17179869653
// MI455X (gfx1250) — matrix operand order repaired and checked by reading, not compiled, not run on hardware
//
#include <hip/hip_runtime.h>

// ---------------------------------------------------------------------------
// DoubleStreamBlock (Flux-style MMDiT) for gfx1250 (MI455X).
// All GEMM contractions run on v_wmma_f32_16x16x32_bf16 (f32 accumulate).
// Norms / softmax / residuals stay in f32.
// Tile staging uses GLOBAL_LOAD_ASYNC_TO_LDS_B128 (ASYNCcnt) when the builtin
// exists, with a VGPR-bounce fallback; GEMM K-tiles are double-buffered so the
// async DMA into buffer cur^1 overlaps WMMA on buffer cur.
// ---------------------------------------------------------------------------

typedef __attribute__((ext_vector_type(16))) __bf16 v16bf;
typedef __attribute__((ext_vector_type(8)))  float  v8f;

struct alignas(16) B128 { unsigned int a, b, c, d; };
union Frag { B128 u[2]; v16bf v; };

static constexpr int Dm   = 1024;   // model dim
static constexpr int Lo   = 3072;   // obs tokens
static constexpr int Lc   = 1024;   // cond tokens
static constexpr int LT   = 4096;   // total tokens (attention order: cond then obs)
static constexpr int NH   = 16;     // heads
static constexpr int DH   = 64;     // head dim
static constexpr int MH   = 4096;   // mlp hidden

#if defined(__has_builtin)
#  if __has_builtin(__builtin_amdgcn_global_load_async_to_lds_b128)
#    define HAVE_ASYNC_LDS 1
#  endif
#endif
#ifndef HAVE_ASYNC_LDS
#  define HAVE_ASYNC_LDS 0
#endif

#if HAVE_ASYNC_LDS
// Builtin signature (from hipcc diagnostic): parameter is a pointer to a
// 16-byte int vector; AS1 (global) source first, AS3 (LDS) dest second.
typedef int v4i_vs __attribute__((__vector_size__(16)));
typedef __attribute__((address_space(1))) v4i_vs g_v4i;
typedef __attribute__((address_space(3))) v4i_vs l_v4i;
__device__ __forceinline__ void async_copy16(const void* g, void* l) {
  // 16B global -> LDS, tracked by ASYNCcnt, bypasses VGPRs
  __builtin_amdgcn_global_load_async_to_lds_b128((g_v4i*)g, (l_v4i*)l, 0, 0);
}
__device__ __forceinline__ void async_wait0() {
#if __has_builtin(__builtin_amdgcn_s_wait_asynccnt)
  __builtin_amdgcn_s_wait_asynccnt(0);
#else
  asm volatile("s_wait_asynccnt 0x0" ::: "memory");
#endif
}
#endif

__device__ __forceinline__ unsigned short f2bf(float x) {
  unsigned int u = __float_as_uint(x);
  u += 0x7FFFu + ((u >> 16) & 1u);
  return (unsigned short)(u >> 16);
}

__device__ __forceinline__ v8f wmma_bf16(v16bf a, v16bf b, v8f c) {
  return __builtin_amdgcn_wmma_f32_16x16x32_bf16(false, a, false, b, (short)0, c,
                                                 false, false);
}

// ---------------------------------------------------------------------------
// silu(vec) -> ws  (1024 elems)
// ---------------------------------------------------------------------------
__global__ void k_silu(const float* __restrict__ vec, float* __restrict__ out) {
  int i = blockIdx.x * 256 + threadIdx.x;
  if (i < Dm) {
    float v = vec[i];
    out[i] = v / (1.0f + __expf(-v));
  }
}

// ---------------------------------------------------------------------------
// modulation: out[s][0:6144] = silu(vec) @ mod_w[s] + mod_b[s]
// ---------------------------------------------------------------------------
__global__ void k_modvec(const float* __restrict__ sv,
                         const float* __restrict__ ow, const float* __restrict__ ob,
                         const float* __restrict__ cw, const float* __restrict__ cb,
                         float* __restrict__ out) {
  int j = blockIdx.x * 256 + threadIdx.x;      // 0..12287
  int s = j / 6144;
  int jj = j - s * 6144;
  const float* w = s ? cw : ow;
  const float* b = s ? cb : ob;
  float acc = b[jj];
  for (int i = 0; i < Dm; ++i)
    acc = fmaf(sv[i], w[(size_t)i * 6144 + jj], acc);
  out[j] = acc;
}

// ---------------------------------------------------------------------------
// fp32 (K x N) -> bf16 transposed (N x K), tiled through LDS
// ---------------------------------------------------------------------------
__global__ __launch_bounds__(256) void k_tcvt(const float* __restrict__ W,
                                              unsigned short* __restrict__ Wt,
                                              int K, int N) {
  __shared__ float tile[32][33];
  int nb = blockIdx.x * 32, kb = blockIdx.y * 32;
  int tx = threadIdx.x, ty = threadIdx.y;
#pragma unroll
  for (int r = 0; r < 4; ++r)
    tile[ty + r * 8][tx] = W[(size_t)(kb + ty + r * 8) * N + nb + tx];
  __syncthreads();
#pragma unroll
  for (int r = 0; r < 4; ++r)
    Wt[(size_t)(nb + ty + r * 8) * K + kb + tx] = f2bf(tile[tx][ty + r * 8]);
}

// ---------------------------------------------------------------------------
// LayerNorm + modulation: y = (1+scale)*ln(x) + shift  (bf16 out)
// Optionally copies x into the residual/output buffer (pass 1).
// One block per row of 1024.
// ---------------------------------------------------------------------------
__global__ __launch_bounds__(256) void k_ln_mod(const float* __restrict__ x,
                                                float* __restrict__ copy_out,
                                                const float* __restrict__ shift,
                                                const float* __restrict__ scale,
                                                unsigned short* __restrict__ y) {
  __shared__ float rs[8], rs2[8];
  int row = blockIdx.x;
  const float* xr = x + (size_t)row * Dm;
  int tid = threadIdx.x, lane = tid & 31, wave = tid >> 5;
  float vals[4], s = 0.f, s2 = 0.f;
#pragma unroll
  for (int i = 0; i < 4; ++i) {
    float v = xr[tid + i * 256];
    vals[i] = v; s += v; s2 += v * v;
  }
#pragma unroll
  for (int m = 1; m < 32; m <<= 1) {
    s  += __shfl_xor(s,  m, 32);
    s2 += __shfl_xor(s2, m, 32);
  }
  if (lane == 0) { rs[wave] = s; rs2[wave] = s2; }
  __syncthreads();
  float S = 0.f, S2 = 0.f;
#pragma unroll
  for (int w = 0; w < 8; ++w) { S += rs[w]; S2 += rs2[w]; }
  float mean = S * (1.0f / Dm);
  float var  = S2 * (1.0f / Dm) - mean * mean;
  float inv  = rsqrtf(var + 1e-6f);
#pragma unroll
  for (int i = 0; i < 4; ++i) {
    int c = tid + i * 256;
    float xh = (vals[i] - mean) * inv;
    float o  = fmaf(scale[c], xh, xh) + shift[c];   // (1+scale)*xh + shift
    y[(size_t)row * Dm + c] = f2bf(o);
    if (copy_out) copy_out[(size_t)row * Dm + c] = vals[i];
  }
}

// ---------------------------------------------------------------------------
// Generic bf16 GEMM, f32 accumulate.  C(MxN) = A(MxK) * Bt(NxK)^T
//   mode 0: outf = acc                         (f32)
//   mode 1: outb = gelu_tanh(acc + bias)       (bf16)
//   mode 2: outf += gate[col]*(acc + bias)     (f32, residual)
// Block tile 128x128, BK=32, 8 waves (4x2), wave tile 32x64 (2x4 WMMA accs).
// Double-buffered LDS; next tile staged via async-to-LDS while WMMA runs.
// ---------------------------------------------------------------------------
__global__ __launch_bounds__(256, 1) void k_gemm(
    const unsigned short* __restrict__ A, const unsigned short* __restrict__ Bt,
    const float* __restrict__ bias, const float* __restrict__ gate,
    float* __restrict__ outf, unsigned short* __restrict__ outb,
    int N, int K, int mode) {
  constexpr int SA = 80, SB = 80;                 // padded LDS row strides (bytes)
  __shared__ unsigned char ldsA[2][128 * SA];
  __shared__ unsigned char ldsB[2][128 * SB];

  const int tid = threadIdx.x;
  const int lane = tid & 31, wave = tid >> 5;
  const int hi = lane >> 4, l15 = lane & 15;
  const int wm = wave & 3, wn = wave >> 2;
  const long m0 = (long)blockIdx.y * 128;
  const long n0 = (long)blockIdx.x * 128;
  const int srow = tid >> 1, shalf = tid & 1;

  v8f zv = {0.f, 0.f, 0.f, 0.f, 0.f, 0.f, 0.f, 0.f};
  v8f acc[2][4];
#pragma unroll
  for (int i = 0; i < 2; ++i)
#pragma unroll
    for (int j = 0; j < 4; ++j) acc[i][j] = zv;

  const unsigned short* Arow = A + (m0 + srow) * (long)K + shalf * 16;
  const unsigned short* Brow = Bt + (n0 + srow) * (long)K + shalf * 16;

  auto stage = [&](int k0, int buf) {
    unsigned char* da = &ldsA[buf][srow * SA + shalf * 32];
    unsigned char* db = &ldsB[buf][srow * SB + shalf * 32];
    const unsigned short* ag = Arow + k0;
    const unsigned short* bg = Brow + k0;
#if HAVE_ASYNC_LDS
    async_copy16(ag,     da);
    async_copy16(ag + 8, da + 16);
    async_copy16(bg,     db);
    async_copy16(bg + 8, db + 16);
#else
    B128 a0 = *(const B128*)ag, a1 = *(const B128*)(ag + 8);
    B128 b0 = *(const B128*)bg, b1 = *(const B128*)(bg + 8);
    *(B128*)da = a0; *(B128*)(da + 16) = a1;
    *(B128*)db = b0; *(B128*)(db + 16) = b1;
#endif
  };
  auto stage_wait = [&]() {
#if HAVE_ASYNC_LDS
    async_wait0();
#endif
    __syncthreads();
  };

  stage(0, 0);
  stage_wait();

  for (int k0 = 0; k0 < K; k0 += 32) {
    const int cur = (k0 >> 5) & 1;
    if (k0 + 32 < K) {
      stage(k0 + 32, cur ^ 1);                    // DMA overlaps WMMA below
      __builtin_prefetch(Arow + k0 + 64, 0, 1);   // pull tile-after-next to L2
      __builtin_prefetch(Brow + k0 + 64, 0, 1);
    }

    // A fragments: lane = row m; halves take K {0-7,16-23} / {8-15,24-31}
    Frag af[2];
#pragma unroll
    for (int sm = 0; sm < 2; ++sm) {
      const unsigned char* p = &ldsA[cur][(wm * 32 + sm * 16 + l15) * SA + hi * 16];
      af[sm].u[0] = *(const B128*)p;
      af[sm].u[1] = *(const B128*)(p + 32);
    }
    // B fragments: lane = col n; halves take K {0-7,16-23} / {8-15,24-31}
    Frag bf[4];
#pragma unroll
    for (int sn = 0; sn < 4; ++sn) {
      const unsigned char* p = &ldsB[cur][(wn * 64 + sn * 16 + l15) * SB + hi * 16];
      bf[sn].u[0] = *(const B128*)p;
      bf[sn].u[1] = *(const B128*)(p + 32);
    }
#pragma unroll
    for (int sm = 0; sm < 2; ++sm)
#pragma unroll
      for (int sn = 0; sn < 4; ++sn)
        acc[sm][sn] = wmma_bf16(af[sm].v, bf[sn].v, acc[sm][sn]);

    stage_wait();   // wmma's dscnt waits ensure reads of `cur` done pre-barrier
  }

#pragma unroll
  for (int sm = 0; sm < 2; ++sm)
#pragma unroll
    for (int sn = 0; sn < 4; ++sn)
#pragma unroll
      for (int r = 0; r < 8; ++r) {
        long rr = m0 + wm * 32 + sm * 16 + hi * 8 + r;
        long cc = n0 + wn * 64 + sn * 16 + l15;
        float v = acc[sm][sn][r];
        if (mode == 0) {
          outf[rr * N + cc] = v;
        } else if (mode == 1) {
          v += bias[cc];
          float g = 0.5f * v *
                    (1.0f + tanhf(0.7978845608f * (v + 0.044715f * v * v * v)));
          outb[rr * N + cc] = f2bf(g);
        } else {
          outf[rr * N + cc] += gate[cc] * (v + bias[cc]);
        }
      }
}

// ---------------------------------------------------------------------------
// QKV post: per-head RMSNorm(q,k) + RoPE, pack into (H, L, DH) bf16 buffers.
// One wave per (t, head); lane owns rope pair (2*lane, 2*lane+1).
// qkv buffer rows are in output order (obs rows 0..3071, cond rows 3072..4095);
// attention index t: t<1024 -> cond, else obs.
// ---------------------------------------------------------------------------
__global__ __launch_bounds__(256) void k_qkv_post(
    const float* __restrict__ qkv, const float* __restrict__ pe,
    const float* __restrict__ oqs, const float* __restrict__ oks,
    const float* __restrict__ cqs, const float* __restrict__ cks,
    unsigned short* __restrict__ Qb, unsigned short* __restrict__ Kb,
    unsigned short* __restrict__ Vb) {
  int gw = (blockIdx.x * 256 + threadIdx.x) >> 5;  // 0..L*H-1
  int lane = threadIdx.x & 31;
  int h = gw & (NH - 1);
  int t = gw >> 4;
  bool is_cond = t < Lc;
  int row = is_cond ? (Lo + t) : (t - Lc);
  const float* qs = is_cond ? cqs : oqs;
  const float* ks = is_cond ? cks : oks;
  const float* base = qkv + (size_t)row * (3 * Dm) + h * DH;
  int d0 = lane * 2, d1 = d0 + 1;
  float q0 = base[d0],        q1 = base[d1];
  float k0 = base[Dm + d0],   k1 = base[Dm + d1];
  float v0 = base[2 * Dm + d0], v1 = base[2 * Dm + d1];
  float sq = q0 * q0 + q1 * q1, sk = k0 * k0 + k1 * k1;
#pragma unroll
  for (int m = 1; m < 32; m <<= 1) {
    sq += __shfl_xor(sq, m, 32);
    sk += __shfl_xor(sk, m, 32);
  }
  float rq = rsqrtf(sq * (1.0f / DH) + 1e-6f);
  float rk = rsqrtf(sk * (1.0f / DH) + 1e-6f);
  q0 *= rq * qs[d0]; q1 *= rq * qs[d1];
  k0 *= rk * ks[d0]; k1 *= rk * ks[d1];
  const float* pp = pe + ((size_t)t * 32 + lane) * 4;
  float e00 = pp[0], e01 = pp[1], e10 = pp[2], e11 = pp[3];
  float qo0 = e00 * q0 + e01 * q1, qo1 = e10 * q0 + e11 * q1;
  float ko0 = e00 * k0 + e01 * k1, ko1 = e10 * k0 + e11 * k1;
  size_t o = ((size_t)h * LT + t) * DH + d0;
  Qb[o] = f2bf(qo0); Qb[o + 1] = f2bf(qo1);
  Kb[o] = f2bf(ko0); Kb[o + 1] = f2bf(ko1);
  Vb[o] = f2bf(v0);  Vb[o + 1] = f2bf(v1);
}

// ---------------------------------------------------------------------------
// Flash attention.  grid = (L/64, H), block = 128 (4 waves).
// Each wave owns a 16-query tile; WG stages 128-key K/V tiles in LDS (K tile
// via async-to-LDS). Inner loop: 32-key chunks -> S (2x v8f) -> online softmax
// -> P (bf16 via LDS transpose to A-layout) -> P@V accumulate (4 WMMA).
// Output written bf16 into (L x Dm) buffer, rows in output order.
// ---------------------------------------------------------------------------
__global__ __launch_bounds__(128, 1) void k_attn(
    const unsigned short* __restrict__ Qb, const unsigned short* __restrict__ Kb,
    const unsigned short* __restrict__ Vb, unsigned short* __restrict__ attn) {
  constexpr int SK = 144;   // K tile row stride (128B data + 16 pad)
  constexpr int SV = 272;   // Vt row stride (256B data + 16 pad)
  constexpr int SP = 80;    // P row stride
  __shared__ unsigned char sK[128 * SK];
  __shared__ unsigned char sVt[64 * SV];
  __shared__ unsigned char sP[4 * 16 * SP];

  int h = blockIdx.y;
  int tid = threadIdx.x, wave = tid >> 5, lane = tid & 31;
  int hi = lane >> 4, l15 = lane & 15;
  int t0 = blockIdx.x * 64 + wave * 16;

  const unsigned short* Qh = Qb + (size_t)h * LT * DH;
  const unsigned short* Kh = Kb + (size_t)h * LT * DH;
  const unsigned short* Vh = Vb + (size_t)h * LT * DH;

  // Q fragments (held in registers for the whole pass)
  Frag qf[2];
  {
    const unsigned char* qrow = (const unsigned char*)(Qh + (size_t)(t0 + l15) * DH);
#pragma unroll
    for (int s = 0; s < 2; ++s) {
      qf[s].u[0] = *(const B128*)(qrow + s * 64 + hi * 16);
      qf[s].u[1] = *(const B128*)(qrow + s * 64 + hi * 16 + 32);
    }
  }

  v8f zv = {0.f, 0.f, 0.f, 0.f, 0.f, 0.f, 0.f, 0.f};
  v8f oacc[4];
#pragma unroll
  for (int i = 0; i < 4; ++i) oacc[i] = zv;
  float mrow[8], lrow[8];
#pragma unroll
  for (int r = 0; r < 8; ++r) { mrow[r] = -1e30f; lrow[r] = 0.f; }

  for (int kb = 0; kb < LT / 128; ++kb) {
    __syncthreads();
    {   // stage K (row-major, async DMA) and V (transposed, via VGPRs)
      int key = tid;  // 0..127
      const unsigned short* kp = Kh + (size_t)(kb * 128 + key) * DH;
      unsigned char* kd = sK + key * SK;
#if HAVE_ASYNC_LDS
      async_copy16(kp,      kd);
      async_copy16(kp + 8,  kd + 16);
      async_copy16(kp + 16, kd + 32);
      async_copy16(kp + 24, kd + 48);
#else
      const B128* kq = (const B128*)kp;
      ((B128*)kd)[0] = kq[0]; ((B128*)(kd + 16))[0] = kq[1];
      ((B128*)(kd + 32))[0] = kq[2]; ((B128*)(kd + 48))[0] = kq[3];
#endif
      const unsigned short* vp = Vh + (size_t)(kb * 128 + key) * DH;
#pragma unroll
      for (int d = 0; d < DH; ++d)
        *(unsigned short*)(sVt + d * SV + key * 2) = vp[d];
    }
#if HAVE_ASYNC_LDS
    async_wait0();
#endif
    __syncthreads();

    for (int c = 0; c < 4; ++c) {
      int lk = c * 32;
      // K fragments (B operand of Q@K^T): lane = key, K-dim runs over DH
      Frag kf[2][2];
#pragma unroll
      for (int ns = 0; ns < 2; ++ns)
#pragma unroll
        for (int ksu = 0; ksu < 2; ++ksu) {
          const unsigned char* p = sK + (lk + ns * 16 + l15) * SK + ksu * 64 + hi * 16;
          kf[ns][ksu].u[0] = *(const B128*)p;
          kf[ns][ksu].u[1] = *(const B128*)(p + 32);
        }
      v8f s0 = zv, s1 = zv;
      s0 = wmma_bf16(qf[0].v, kf[0][0].v, s0);
      s0 = wmma_bf16(qf[1].v, kf[0][1].v, s0);
      s1 = wmma_bf16(qf[0].v, kf[1][0].v, s1);
      s1 = wmma_bf16(qf[1].v, kf[1][1].v, s1);

      float pr0[8], pr1[8];
#pragma unroll
      for (int r = 0; r < 8; ++r) {
        float a = s0[r] * 0.125f, b = s1[r] * 0.125f;   // Dh^-0.5
        float mx = fmaxf(a, b);
#pragma unroll
        for (int m = 1; m < 16; m <<= 1) mx = fmaxf(mx, __shfl_xor(mx, m, 32));
        float mnew = fmaxf(mrow[r], mx);
        float p0 = __expf(a - mnew), p1 = __expf(b - mnew);
        float sum = p0 + p1;
#pragma unroll
        for (int m = 1; m < 16; m <<= 1) sum += __shfl_xor(sum, m, 32);
        float f = __expf(mrow[r] - mnew);
        lrow[r] = lrow[r] * f + sum;
        mrow[r] = mnew;
#pragma unroll
        for (int ns = 0; ns < 4; ++ns) oacc[ns][r] *= f;
        pr0[r] = p0; pr1[r] = p1;
      }

      // P: C-layout f32 -> A-layout bf16 via per-wave LDS transpose
      unsigned char* pbase = sP + wave * 16 * SP;
#pragma unroll
      for (int r = 0; r < 8; ++r) {
        int m = hi * 8 + r;
        *(unsigned short*)(pbase + m * SP + l15 * 2)        = f2bf(pr0[r]);
        *(unsigned short*)(pbase + m * SP + 32 + l15 * 2)   = f2bf(pr1[r]);
      }
      __syncthreads();
      Frag pf;
      {
        const unsigned char* p = pbase + l15 * SP + hi * 16;
        pf.u[0] = *(const B128*)p;
        pf.u[1] = *(const B128*)(p + 32);
      }
      // V fragments (B operand of P@V): lane = dim d, K runs over keys
#pragma unroll
      for (int ns = 0; ns < 4; ++ns) {
        Frag vf;
        const unsigned char* p = sVt + (ns * 16 + l15) * SV + (lk + hi * 8) * 2;
        vf.u[0] = *(const B128*)p;
        vf.u[1] = *(const B128*)(p + 32);
        oacc[ns] = wmma_bf16(pf.v, vf.v, oacc[ns]);
      }
    }
  }

  // epilogue: O /= l, store bf16 to (L x Dm) buffer in output row order
#pragma unroll
  for (int ns = 0; ns < 4; ++ns)
#pragma unroll
    for (int r = 0; r < 8; ++r) {
      int t = t0 + hi * 8 + r;
      int orow = (t < Lc) ? (Lo + t) : (t - Lc);
      float v = oacc[ns][r] / lrow[r];
      attn[(size_t)orow * Dm + h * DH + ns * 16 + l15] = f2bf(v);
    }
}

// ---------------------------------------------------------------------------
// Host orchestration
// ---------------------------------------------------------------------------
extern "C" void kernel_launch(void* const* d_in, const int* in_sizes, int n_in,
                              void* d_out, int out_size, void* d_ws, size_t ws_size,
                              hipStream_t stream) {
  (void)in_sizes; (void)n_in; (void)out_size; (void)ws_size;
  const float* obs        = (const float*)d_in[0];
  const float* cond       = (const float*)d_in[1];
  const float* vec        = (const float*)d_in[2];
  const float* pe         = (const float*)d_in[3];
  const float* mod_w[2]   = {(const float*)d_in[4],  (const float*)d_in[15]};
  const float* mod_b[2]   = {(const float*)d_in[5],  (const float*)d_in[16]};
  const float* qkv_w[2]   = {(const float*)d_in[6],  (const float*)d_in[17]};
  const float* q_scale[2] = {(const float*)d_in[7],  (const float*)d_in[18]};
  const float* k_scale[2] = {(const float*)d_in[8],  (const float*)d_in[19]};
  const float* proj_w[2]  = {(const float*)d_in[9],  (const float*)d_in[20]};
  const float* proj_b[2]  = {(const float*)d_in[10], (const float*)d_in[21]};
  const float* mlp_w1[2]  = {(const float*)d_in[11], (const float*)d_in[22]};
  const float* mlp_b1[2]  = {(const float*)d_in[12], (const float*)d_in[23]};
  const float* mlp_w2[2]  = {(const float*)d_in[13], (const float*)d_in[24]};
  const float* mlp_b2[2]  = {(const float*)d_in[14], (const float*)d_in[25]};
  float* out = (float*)d_out;   // obs rows 0..3071, cond rows 3072..4095
  unsigned char* ws = (unsigned char*)d_ws;

  size_t off = 0;
  auto alloc = [&](size_t bytes) {
    size_t o = off; off += (bytes + 255) & ~(size_t)255; return o;
  };
  float*          silu   = (float*)(ws + alloc((size_t)Dm * 4));
  float*          modv   = (float*)(ws + alloc((size_t)2 * 6144 * 4));
  unsigned short* qkvwt  = (unsigned short*)(ws + alloc((size_t)2 * 3 * Dm * Dm * 2));
  unsigned short* projwt = (unsigned short*)(ws + alloc((size_t)2 * Dm * Dm * 2));
  unsigned short* mlp1wt = (unsigned short*)(ws + alloc((size_t)2 * MH * Dm * 2));
  unsigned short* mlp2wt = (unsigned short*)(ws + alloc((size_t)2 * Dm * MH * 2));
  unsigned short* xmod   = (unsigned short*)(ws + alloc((size_t)LT * Dm * 2));
  float*          qkvf   = (float*)(ws + alloc((size_t)LT * 3 * Dm * 4));
  unsigned short* Qb     = (unsigned short*)(ws + alloc((size_t)NH * LT * DH * 2));
  unsigned short* Kb     = (unsigned short*)(ws + alloc((size_t)NH * LT * DH * 2));
  unsigned short* Vb     = (unsigned short*)(ws + alloc((size_t)NH * LT * DH * 2));
  unsigned short* attnb  = (unsigned short*)(ws + alloc((size_t)LT * Dm * 2));
  unsigned short* hid    = (unsigned short*)(ws + alloc((size_t)LT * MH * 2));

  const long RB[2] = {0, Lo};          // row bases in output order
  const int  Ms[2] = {Lo, Lc};
  const float* x_in[2] = {obs, cond};

  auto gemm = [&](const unsigned short* A, const unsigned short* Bt,
                  const float* bias, const float* gate, float* of,
                  unsigned short* ob, int M, int N, int K, int mode) {
    dim3 g(N / 128, M / 128);
    k_gemm<<<g, 256, 0, stream>>>(A, Bt, bias, gate, of, ob, N, K, mode);
  };

  // 1. modulation vectors
  k_silu<<<4, 256, 0, stream>>>(vec, silu);
  k_modvec<<<48, 256, 0, stream>>>(silu, mod_w[0], mod_b[0], mod_w[1], mod_b[1], modv);

  // 2. weight conversion (fp32 KxN -> bf16 NxK)
  for (int s = 0; s < 2; ++s) {
    k_tcvt<<<dim3(3 * Dm / 32, Dm / 32), dim3(32, 8), 0, stream>>>(
        qkv_w[s], qkvwt + (size_t)s * 3 * Dm * Dm, Dm, 3 * Dm);
    k_tcvt<<<dim3(Dm / 32, Dm / 32), dim3(32, 8), 0, stream>>>(
        proj_w[s], projwt + (size_t)s * Dm * Dm, Dm, Dm);
    k_tcvt<<<dim3(MH / 32, Dm / 32), dim3(32, 8), 0, stream>>>(
        mlp_w1[s], mlp1wt + (size_t)s * MH * Dm, Dm, MH);
    k_tcvt<<<dim3(Dm / 32, MH / 32), dim3(32, 8), 0, stream>>>(
        mlp_w2[s], mlp2wt + (size_t)s * Dm * MH, MH, Dm);
  }

  // 3. LN + mod1 (also seeds d_out with the residual stream)
  for (int s = 0; s < 2; ++s)
    k_ln_mod<<<Ms[s], 256, 0, stream>>>(x_in[s], out + RB[s] * Dm,
                                        modv + s * 6144 + 0,       // shift1
                                        modv + s * 6144 + 1024,    // scale1
                                        xmod + RB[s] * Dm);

  // 4. QKV GEMM
  for (int s = 0; s < 2; ++s)
    gemm(xmod + RB[s] * Dm, qkvwt + (size_t)s * 3 * Dm * Dm, nullptr, nullptr,
         qkvf + RB[s] * 3 * Dm, nullptr, Ms[s], 3 * Dm, Dm, 0);

  // 5. RMSNorm + RoPE + head packing
  k_qkv_post<<<LT * NH * 32 / 256, 256, 0, stream>>>(
      qkvf, pe, q_scale[0], k_scale[0], q_scale[1], k_scale[1], Qb, Kb, Vb);

  // 6. attention
  k_attn<<<dim3(LT / 64, NH), 128, 0, stream>>>(Qb, Kb, Vb, attnb);

  // 7. proj + gated residual
  for (int s = 0; s < 2; ++s)
    gemm(attnb + RB[s] * Dm, projwt + (size_t)s * Dm * Dm, proj_b[s],
         modv + s * 6144 + 2048,                         // gate1
         out + RB[s] * Dm, nullptr, Ms[s], Dm, Dm, 2);

  // 8. LN + mod2
  for (int s = 0; s < 2; ++s)
    k_ln_mod<<<Ms[s], 256, 0, stream>>>(out + RB[s] * Dm, nullptr,
                                        modv + s * 6144 + 3072,    // shift2
                                        modv + s * 6144 + 4096,    // scale2
                                        xmod + RB[s] * Dm);

  // 9. MLP1 (+bias, gelu) -> bf16 hidden
  for (int s = 0; s < 2; ++s)
    gemm(xmod + RB[s] * Dm, mlp1wt + (size_t)s * MH * Dm, mlp_b1[s], nullptr,
         nullptr, hid + RB[s] * MH, Ms[s], MH, Dm, 1);

  // 10. MLP2 + gated residual
  for (int s = 0; s < 2; ++s)
    gemm(hid + RB[s] * MH, mlp2wt + (size_t)s * Dm * MH, mlp_b2[s],
         modv + s * 6144 + 5120,                         // gate2
         out + RB[s] * Dm, nullptr, Ms[s], Dm, MH, 2);
}
